// InstanceLoss_70927089926677
// MI455X (gfx1250) — matrix operand order repaired and checked by reading, not compiled, not run on hardware
//
#include <hip/hip_runtime.h>
#include <hip/hip_bf16.h>

typedef __attribute__((ext_vector_type(16))) _Float16 v16h;
typedef __attribute__((ext_vector_type(8)))  _Float16 v8h;
typedef __attribute__((ext_vector_type(8)))  float    v8f;

#define B_ROWS 4096
#define D_DIM  256
#define N_ROWS 8192
#define INV_T  2.0f            // 1 / TEMPERATURE
#define COL_CHUNKS 8
#define ROWS_PER_BLOCK 128     // 8 waves x 16 rows
#define COLS_PER_CHUNK (N_ROWS / COL_CHUNKS)   // 1024 columns per workgroup
#define TILES 4                // column tiles in flight (64 cols / outer step)

// -----------------------------------------------------------------------------
// Kernel 1: L2-normalize both views, emit f16 reps[N,256], f32 positives[B].
// One wave (32 lanes) per row; each lane owns 8 contiguous floats.
// -----------------------------------------------------------------------------
__global__ __launch_bounds__(256)
void nrm_pos_kernel(const float* __restrict__ ei, const float* __restrict__ ej,
                    _Float16* __restrict__ reps, float* __restrict__ pos) {
    const int wave = threadIdx.x >> 5;
    const int lane = threadIdx.x & 31;
    const int row  = blockIdx.x * 8 + wave;          // 0..4095

    const float* pi = ei + row * D_DIM + lane * 8;
    const float* pj = ej + row * D_DIM + lane * 8;
    float xi[8], xj[8];
    float sii = 0.f, sjj = 0.f, sij = 0.f;
#pragma unroll
    for (int k = 0; k < 8; ++k) {
        xi[k] = pi[k];
        xj[k] = pj[k];
        sii += xi[k] * xi[k];
        sjj += xj[k] * xj[k];
        sij += xi[k] * xj[k];
    }
#pragma unroll
    for (int m = 16; m; m >>= 1) {
        sii += __shfl_xor(sii, m, 32);
        sjj += __shfl_xor(sjj, m, 32);
        sij += __shfl_xor(sij, m, 32);
    }
    const float ni = 1.0f / fmaxf(sqrtf(sii), 1e-12f);
    const float nj = 1.0f / fmaxf(sqrtf(sjj), 1e-12f);

    v8h qi, qj;
#pragma unroll
    for (int k = 0; k < 8; ++k) {
        qi[k] = (_Float16)(xi[k] * ni);
        qj[k] = (_Float16)(xj[k] * nj);
    }
    *(v8h*)(reps + (size_t)row * D_DIM + lane * 8)            = qi;
    *(v8h*)(reps + (size_t)(row + B_ROWS) * D_DIM + lane * 8) = qj;

    if (lane == 0) pos[row] = sij * ni * nj;   // exact-f32 positive cosine
}

// -----------------------------------------------------------------------------
// Kernel 2: fused sim-GEMM + masked exp row-sum, software-pipelined.
// grid = 512 wgs: blockIdx.x>>3 = row block (128 rows), &7 = column chunk.
// Per wave: A panel (16x256 f16) resident in 64 VGPRs.  4 column tiles in
// flight with ping-pong B buffers; per K-stage the schedule is pinned with
// sched_group_barrier to [8 VMEM reads (stage kc+1)] -> [4 WMMAs (stage kc)],
// so the wait before each WMMA block is partial and global_load_b128 latency
// overlaps v_wmma issue.  Diagonal masked exactly; deterministic stores.
// -----------------------------------------------------------------------------
__global__ __launch_bounds__(256)
void simsum_kernel(const _Float16* __restrict__ reps, float* __restrict__ partial) {
    const int lane    = threadIdx.x & 31;
    const int wave    = threadIdx.x >> 5;
    const int lane15  = lane & 15;
    const int laneHi  = lane >> 4;
    const int rowBlk  = blockIdx.x >> 3;
    const int colChnk = blockIdx.x & 7;
    const int mBase   = rowBlk * ROWS_PER_BLOCK + wave * 16;

    // ---- load A fragments: a[kc] = 16x32 f16 tile (K chunk kc) ----
    v16h a[8];
    {
        const _Float16* arow = reps + (size_t)(mBase + lane15) * D_DIM + laneHi * 16;
#pragma unroll
        for (int kc = 0; kc < 8; ++kc) {
            v8h c0 = *(const v8h*)(arow + kc * 32);        // K 0..7  (per half-wave)
            v8h c1 = *(const v8h*)(arow + kc * 32 + 8);    // K 8..15
            v16h av;
#pragma unroll
            for (int i = 0; i < 8; ++i) { av[i] = c0[i]; av[i + 8] = c1[i]; }
            a[kc] = av;
        }
    }

    // per-lane B base: row (col + lane15), 32B chunk selected by half-wave
    const _Float16* bbase = reps + (size_t)lane15 * D_DIM + laneHi * 16;
    const int nChunkBase  = colChnk * COLS_PER_CHUNK;

    v8f acc = {};                                   // per-lane exp row-sums

    // prologue: group 0, kc = 0 fragments into buffer 0
    v16h b[2][TILES];
#pragma unroll
    for (int t = 0; t < TILES; ++t)
        b[0][t] = *(const v16h*)(bbase + (size_t)(nChunkBase + t * 16) * D_DIM);

    for (int ct = 0; ct < COLS_PER_CHUNK / 16; ct += TILES) {
        const int n0 = nChunkBase + ct * 16;
        // next group's base (wrap keeps the speculative prefetch in-bounds)
        const int nn = nChunkBase + (((ct + TILES) & (COLS_PER_CHUNK / 16 - 1)) * 16);

        v8f c[TILES];
#pragma unroll
        for (int t = 0; t < TILES; ++t) c[t] = (v8f){};

#pragma unroll
        for (int kc = 0; kc < 8; ++kc) {
            const int cur = kc & 1, nxt = cur ^ 1;
            // issue next stage's loads (kc==7 prefetches next group's kc=0)
#pragma unroll
            for (int t = 0; t < TILES; ++t) {
                const _Float16* p = (kc < 7)
                    ? bbase + (size_t)(n0 + t * 16) * D_DIM + (kc + 1) * 32
                    : bbase + (size_t)(nn + t * 16) * D_DIM;
                b[nxt][t] = *(const v16h*)p;
            }
            // 4 independent WMMA chains on the current buffer
#pragma unroll
            for (int t = 0; t < TILES; ++t)
                c[t] = __builtin_amdgcn_wmma_f32_16x16x32_f16(false, a[kc], false,
                                                              b[cur][t], (short)0,
                                                              c[t], false, false);
            // pin the pipeline: 8 VMEM reads then 4 MFMA/WMMA per stage
            __builtin_amdgcn_sched_group_barrier(0x020, 8, 0);  // VMEM read x8
            __builtin_amdgcn_sched_group_barrier(0x008, 4, 0);  // WMMA x4
        }

        // epilogue: exp + diagonal mask + register row-sum (overlaps prefetch)
        // C layout: VGPR v, lanes 0-15 -> (M=v, N=lane); lanes 16-31 -> (M=v+8, N=lane-16)
#pragma unroll
        for (int v = 0; v < 8; ++v) {
            const int gm = mBase + v + laneHi * 8;
            float s = 0.0f;
#pragma unroll
            for (int t = 0; t < TILES; ++t) {
                float e = __expf(INV_T * c[t][v]);
                if (gm == n0 + t * 16 + lane15) e = 0.0f;
                s += e;
            }
            acc[v] += s;
        }
    }

    // reduce across the 16 lanes holding one row (xor masks stay within halves)
#pragma unroll
    for (int v = 0; v < 8; ++v) {
        float s = acc[v];
        s += __shfl_xor(s, 1, 32);
        s += __shfl_xor(s, 2, 32);
        s += __shfl_xor(s, 4, 32);
        s += __shfl_xor(s, 8, 32);
        acc[v] = s;
    }
    if (lane15 == 0) {
#pragma unroll
        for (int v = 0; v < 8; ++v)
            partial[(size_t)colChnk * N_ROWS + mBase + v + laneHi * 8] = acc[v];
    }
}

// -----------------------------------------------------------------------------
// Kernel 3: loss = (1/N) * sum_r [ log(denom_r) - pos_{r mod B} / T ]
// -----------------------------------------------------------------------------
__global__ __launch_bounds__(256)
void loss_reduce_kernel(const float* __restrict__ partial, const float* __restrict__ pos,
                        float* __restrict__ out) {
    __shared__ float sdata[8];
    float acc = 0.0f;
    for (int r = threadIdx.x; r < N_ROWS; r += 256) {
        float denom = 0.0f;
#pragma unroll
        for (int cc = 0; cc < COL_CHUNKS; ++cc) denom += partial[(size_t)cc * N_ROWS + r];
        acc += __logf(denom) - INV_T * pos[r & (B_ROWS - 1)];
    }
#pragma unroll
    for (int m = 16; m; m >>= 1) acc += __shfl_xor(acc, m, 32);
    const int lane = threadIdx.x & 31, wave = threadIdx.x >> 5;
    if (lane == 0) sdata[wave] = acc;
    __syncthreads();
    if (threadIdx.x == 0) {
        float t = 0.0f;
#pragma unroll
        for (int w = 0; w < 8; ++w) t += sdata[w];
        out[0] = t / (float)N_ROWS;
    }
}

// -----------------------------------------------------------------------------
// Launch
// -----------------------------------------------------------------------------
extern "C" void kernel_launch(void* const* d_in, const int* in_sizes, int n_in,
                              void* d_out, int out_size, void* d_ws, size_t ws_size,
                              hipStream_t stream) {
    const float* ei = (const float*)d_in[0];   // [4096,1,256] f32
    const float* ej = (const float*)d_in[1];   // [4096,1,256] f32

    // workspace layout: reps(f16, 4MB) | pos(f32, 16KB) | partial(f32, 256KB)
    _Float16* reps   = (_Float16*)d_ws;
    float*    pos    = (float*)((char*)d_ws + (size_t)N_ROWS * D_DIM * sizeof(_Float16));
    float*    part   = pos + B_ROWS;

    nrm_pos_kernel<<<B_ROWS / 8, 256, 0, stream>>>(ei, ej, reps, pos);
    simsum_kernel<<<(N_ROWS / ROWS_PER_BLOCK) * COL_CHUNKS, 256, 0, stream>>>(reps, part);
    loss_reduce_kernel<<<1, 256, 0, stream>>>(part, pos, (float*)d_out);
}
